// PCProj_24017457119740
// MI455X (gfx1250) — compile-verified, matrix operand order checked by reading, not run on hardware
//
#include <hip/hip_runtime.h>
#include <math.h>

// ---------------- types for WMMA ----------------
typedef __bf16 bf16_t;
typedef __attribute__((ext_vector_type(16))) bf16_t v16bf;
typedef __attribute__((ext_vector_type(8)))  float  v8f;

union FragBF {
    v16bf v;
    unsigned int u[8];
    uint4 q[2];
};

static __device__ inline v8f wmma_bf16(v16bf a, v16bf b, v8f c) {
    // D = A(16x32 bf16) * B(32x16 bf16) + C(16x16 f32)
    return __builtin_amdgcn_wmma_f32_16x16x32_bf16(false, a, false, b, (short)0, c,
                                                   false, false);
}

static __device__ inline unsigned int bf16_bits(float x) {
    unsigned int u = __float_as_uint(x);
    unsigned int r = u + 0x7FFFu + ((u >> 16) & 1u);   // round-to-nearest-even
    return r >> 16;
}
static __device__ inline unsigned int bf16_pack2(float lo, float hi) {
    return bf16_bits(lo) | (bf16_bits(hi) << 16);
}

// klocal mapping for 16-bit A/B fragments (ISA 7.12.2, 16-bit A 16x32):
// lanes 0-15: v0-3 -> K 0..7 ; v4-7 -> K 16..23
// lanes16-31: v0-3 -> K 8..15; v4-7 -> K 24..31
static __device__ inline int klocal_of(int v, int half, int laneHi) {
    int base = (v < 4) ? (laneHi ? 8 : 0) : (laneHi ? 24 : 16);
    int vv = (v < 4) ? v : (v - 4);
    return base + 2 * vv + half;
}

// ---------------- problem constants ----------------
#define Bn   8
#define Npt  2048
#define Kk   8
#define OBJ  224
#define XT   14                         // 224/16 tiles along x

// ---------------- workspace layout (bytes) ----------------
#define OFF_GEOM   ((size_t)0)                        // B*8 f32
#define OFF_GNSUM  ((size_t)256)                      // B*4*2 f32
#define OFF_GNSTAT ((size_t)512)                      // B*4*2 f32
#define OFF_F0     ((size_t)1024)                     // B*N*8 f32
#define OFF_KNN    (OFF_F0  + (size_t)Bn*Npt*8*4)     // B*N*8 i32
#define OFF_G      (OFF_KNN + (size_t)Bn*Npt*8*4)     // B*N*K*64 f32
#define OFF_RES    (OFF_G   + (size_t)Bn*Npt*Kk*64*4) // B*224*224*64 f32 (NHWC)
#define OFF_H1     (OFF_RES + (size_t)Bn*OBJ*OBJ*64*4)// bf16 NHWC
#define OFF_H2     (OFF_H1  + (size_t)Bn*OBJ*OBJ*64*2)
#define OFF_P1     (OFF_H2  + (size_t)Bn*OBJ*OBJ*64*2)// 18*4*32*32 B
#define OFF_P2     (OFF_P1  + (size_t)18*4*32*32)
#define OFF_PB     (OFF_P2  + (size_t)18*4*32*32)     // 2*4*32*32 B

// ---------------- zero fill ----------------
__global__ void k_zero(float4* p, long long n4) {
    long long i = (long long)blockIdx.x * blockDim.x + threadIdx.x;
    if (i < n4) p[i] = make_float4(0.f, 0.f, 0.f, 0.f);
}

// ---------------- per-batch geometry ----------------
static __device__ inline float blkRedMin(float v, float* s) {
    int tid = threadIdx.x; s[tid] = v; __syncthreads();
    for (int k = 128; k > 0; k >>= 1) { if (tid < k) s[tid] = fminf(s[tid], s[tid + k]); __syncthreads(); }
    float r = s[0]; __syncthreads(); return r;
}
static __device__ inline float blkRedMax(float v, float* s) {
    int tid = threadIdx.x; s[tid] = v; __syncthreads();
    for (int k = 128; k > 0; k >>= 1) { if (tid < k) s[tid] = fmaxf(s[tid], s[tid + k]); __syncthreads(); }
    float r = s[0]; __syncthreads(); return r;
}

__global__ void k_geom(const float* __restrict__ pts, float* __restrict__ geom) {
    __shared__ float sbuf[256];
    __shared__ float sg[3];
    int b = blockIdx.x, tid = threadIdx.x;
    const float* pb = pts + (size_t)b * Npt * 3;
    float mnx = 1e30f, mny = 1e30f, mxx = -1e30f, mxy = -1e30f;
    for (int n = tid; n < Npt; n += 256) {
        float x = pb[n * 3 + 0], y = pb[n * 3 + 1];
        mnx = fminf(mnx, x); mxx = fmaxf(mxx, x);
        mny = fminf(mny, y); mxy = fmaxf(mxy, y);
    }
    mnx = blkRedMin(mnx, sbuf); mxx = blkRedMax(mxx, sbuf);
    mny = blkRedMin(mny, sbuf); mxy = blkRedMax(mxy, sbuf);
    if (tid == 0) { sg[0] = fmaxf(mxx - mnx, mxy - mny) / (float)(OBJ - 3); sg[1] = mnx; sg[2] = mny; }
    __syncthreads();
    float gridv = sg[0], pmx = sg[1], pmy = sg[2];
    float gmnx = 1e30f, gmny = 1e30f, gmxx = -1e30f, gmxy = -1e30f;
    for (int n = tid; n < Npt; n += 256) {
        float gx = floorf((pb[n * 3 + 0] - pmx) / gridv);
        float gy = floorf((pb[n * 3 + 1] - pmy) / gridv);
        gmnx = fminf(gmnx, gx); gmxx = fmaxf(gmxx, gx);
        gmny = fminf(gmny, gy); gmxy = fmaxf(gmxy, gy);
    }
    gmnx = blkRedMin(gmnx, sbuf); gmxx = blkRedMax(gmxx, sbuf);
    gmny = blkRedMin(gmny, sbuf); gmxy = blkRedMax(gmxy, sbuf);
    if (tid == 0) {
        // dense max = max(idx)+2, dense min = min(idx); center = floor((max+min)/2)
        float cx = floorf((gmxx + 2.f + gmnx) * 0.5f);
        float cy = floorf((gmxy + 2.f + gmny) * 0.5f);
        geom[b * 8 + 0] = pmx;
        geom[b * 8 + 1] = pmy;
        geom[b * 8 + 2] = gridv;
        geom[b * 8 + 3] = (float)(OBJ / 2) - cx - 1.f;
        geom[b * 8 + 4] = (float)(OBJ / 2) - cy - 1.f;
    }
}

// ---------------- KNN (k=8) + input linear f0 ----------------
__global__ void k_knn(const float* __restrict__ pts, const float* __restrict__ w_in,
                      const float* __restrict__ b_in, float* __restrict__ f0,
                      int* __restrict__ knn) {
    __shared__ float sx[256], sy[256], sz[256];
    int tid = threadIdx.x;
    int b = blockIdx.x >> 3;
    int n = ((blockIdx.x & 7) << 8) + tid;
    const float* pb = pts + (size_t)b * Npt * 3;
    float px = pb[n * 3 + 0], py = pb[n * 3 + 1], pz = pb[n * 3 + 2];
    float bd[8]; int bi[8];
#pragma unroll
    for (int i = 0; i < 8; ++i) { bd[i] = 3.4e38f; bi[i] = 0; }
    for (int j0 = 0; j0 < Npt; j0 += 256) {
        sx[tid] = pb[(j0 + tid) * 3 + 0];
        sy[tid] = pb[(j0 + tid) * 3 + 1];
        sz[tid] = pb[(j0 + tid) * 3 + 2];
        __syncthreads();
        for (int jj = 0; jj < 256; ++jj) {
            float dx = px - sx[jj], dy = py - sy[jj], dz = pz - sz[jj];
            float d = dx * dx + dy * dy + dz * dz;
            if (d < bd[7]) {
                float cd = d; int ci = j0 + jj;
#pragma unroll
                for (int s = 0; s < 8; ++s) {
                    if (cd < bd[s]) {
                        float td = bd[s]; int ti = bi[s];
                        bd[s] = cd; bi[s] = ci; cd = td; ci = ti;
                    }
                }
            }
        }
        __syncthreads();
    }
    size_t base = ((size_t)b * Npt + n) * 8;
#pragma unroll
    for (int i = 0; i < 8; ++i) knn[base + i] = bi[i];
#pragma unroll
    for (int o = 0; o < 8; ++o)
        f0[base + o] = px * w_in[o * 3 + 0] + py * w_in[o * 3 + 1] + pz * w_in[o * 3 + 2] + b_in[o];
}

// ---------------- graph GEMM (16->64) + GN sums ----------------
__global__ void k_graph(const float* __restrict__ f0, const int* __restrict__ knn,
                        const float* __restrict__ w_graph, float* __restrict__ g,
                        float* __restrict__ gnsum) {
    __shared__ float ssum[8];
    int tid = threadIdx.x;
    if (tid < 8) ssum[tid] = 0.f;
    __syncthreads();
    size_t r = (size_t)blockIdx.x * 256 + tid;          // < B*N*K
    int b = (int)(r >> 14);
    int rem = (int)(r & 16383);
    int n = rem >> 3, k = rem & 7;
    int j = knn[((size_t)b * Npt + n) * 8 + k];
    const float* fn = f0 + ((size_t)b * Npt + j) * 8;
    const float* fc = f0 + ((size_t)b * Npt + n) * 8;
    float feat[16];
#pragma unroll
    for (int c = 0; c < 8; ++c) { feat[c] = fn[c] - fc[c]; feat[8 + c] = fc[c]; }
    float* go = g + r * 64;
#pragma unroll
    for (int grp = 0; grp < 4; ++grp) {
        float s = 0.f, q = 0.f;
        for (int oo = 0; oo < 16; ++oo) {
            int o = grp * 16 + oo;
            float acc = 0.f;
#pragma unroll
            for (int c = 0; c < 16; ++c) acc += feat[c] * w_graph[o * 16 + c];
            go[o] = acc; s += acc; q += acc * acc;
        }
        atomicAdd(&ssum[grp * 2 + 0], s);
        atomicAdd(&ssum[grp * 2 + 1], q);
    }
    __syncthreads();
    if (tid < 8) atomicAdd(&gnsum[b * 8 + tid], ssum[tid]);
}

__global__ void k_gnfin(const float* __restrict__ gnsum, float* __restrict__ gnstat) {
    int tid = threadIdx.x;
    if (tid < 32) {
        int b = tid >> 2, grp = tid & 3;
        const float cnt = (float)(Npt * Kk * 16);
        float mu = gnsum[b * 8 + grp * 2 + 0] / cnt;
        float var = gnsum[b * 8 + grp * 2 + 1] / cnt - mu * mu;
        gnstat[b * 8 + grp * 2 + 0] = mu;
        gnstat[b * 8 + grp * 2 + 1] = rsqrtf(var + 1e-5f);
    }
}

// ---------------- GN+LeakyReLU+max_k+proj, then scatter-sum ----------------
__global__ void k_fproj(const float* __restrict__ g, const float* __restrict__ gnstat,
                        const float* __restrict__ gn_g, const float* __restrict__ gn_b,
                        const float* __restrict__ w_proj, const float* __restrict__ b_proj,
                        const float* __restrict__ pts, const float* __restrict__ geom,
                        float* __restrict__ res) {
    size_t idx = (size_t)blockIdx.x * 256 + threadIdx.x;   // < B*N
    int b = (int)(idx >> 11), n = (int)(idx & 2047);
    float gmax[64];
#pragma unroll
    for (int c = 0; c < 64; ++c) gmax[c] = -3.0e38f;
    for (int k = 0; k < Kk; ++k) {
        const float* gr = g + (((size_t)b * Npt + n) * 8 + k) * 64;
#pragma unroll
        for (int c = 0; c < 64; ++c) {
            int grp = c >> 4;
            float mu = gnstat[b * 8 + grp * 2 + 0];
            float rs = gnstat[b * 8 + grp * 2 + 1];
            float v = (gr[c] - mu) * rs * gn_g[c] + gn_b[c];
            v = (v >= 0.f) ? v : 0.2f * v;
            gmax[c] = fmaxf(gmax[c], v);
        }
    }
    // scatter target indices (9 per point)
    float pmx = geom[b * 8 + 0], pmy = geom[b * 8 + 1], gridv = geom[b * 8 + 2];
    float offx = geom[b * 8 + 3], offy = geom[b * 8 + 4];
    float gx = floorf((pts[((size_t)b * Npt + n) * 3 + 0] - pmx) / gridv);
    float gy = floorf((pts[((size_t)b * Npt + n) * 3 + 1] - pmy) / gridv);
    int lin[9];
#pragma unroll
    for (int t = 0; t < 9; ++t) {
        int dxo = t / 3 - 1, dyo = t % 3 - 1;
        float fx = gx + (float)dxo + 1.f + offx;
        float fy = gy + (float)dyo + 1.f + offy;
        fx += (fx < 0.f ? 1.f : 0.f) - (fx > (float)(OBJ - 1) ? 1.f : 0.f);
        fy += (fy < 0.f ? 1.f : 0.f) - (fy > (float)(OBJ - 1) ? 1.f : 0.f);
        lin[t] = ((b * OBJ + (int)fx) * OBJ + (int)fy) * 64;
    }
#pragma unroll 1
    for (int o = 0; o < 64; ++o) {
        float acc = b_proj[o];
#pragma unroll
        for (int c = 0; c < 64; ++c) acc += gmax[c] * w_proj[o * 64 + c];
#pragma unroll
        for (int t = 0; t < 9; ++t) atomicAdd(&res[(size_t)lin[t] + o], acc);
    }
}

// ---------------- weight packing into WMMA B-fragment layout ----------------
static __device__ void pack_conv(const float* __restrict__ w, uint4* __restrict__ out, int id) {
    int lane = id & 31, oct = (id >> 5) & 3, kc = id >> 7;   // kc < 18
    int laneHi = lane >> 4;
    int oc = oct * 16 + (lane & 15);
    unsigned int u[8];
#pragma unroll
    for (int v = 0; v < 8; ++v) {
        float vals[2];
#pragma unroll
        for (int half = 0; half < 2; ++half) {
            int kl = klocal_of(v, half, laneHi);
            int Kg = kc * 32 + kl;
            int tap = Kg >> 6, ic = Kg & 63;
            vals[half] = w[((size_t)oc * 64 + ic) * 9 + tap];
        }
        u[v] = bf16_pack2(vals[0], vals[1]);
    }
    uint4* dst = out + ((size_t)(kc * 4 + oct) * 32 + lane) * 2;
    dst[0] = make_uint4(u[0], u[1], u[2], u[3]);
    dst[1] = make_uint4(u[4], u[5], u[6], u[7]);
}
static __device__ void pack_mat64(const float* __restrict__ w, uint4* __restrict__ out, int id) {
    int lane = id & 31, oct = (id >> 5) & 3, kc = id >> 7;   // kc < 2
    int laneHi = lane >> 4;
    int oc = oct * 16 + (lane & 15);
    unsigned int u[8];
#pragma unroll
    for (int v = 0; v < 8; ++v) {
        float vals[2];
#pragma unroll
        for (int half = 0; half < 2; ++half) {
            int ic = kc * 32 + klocal_of(v, half, laneHi);
            vals[half] = w[(size_t)oc * 64 + ic];
        }
        u[v] = bf16_pack2(vals[0], vals[1]);
    }
    uint4* dst = out + ((size_t)(kc * 4 + oct) * 32 + lane) * 2;
    dst[0] = make_uint4(u[0], u[1], u[2], u[3]);
    dst[1] = make_uint4(u[4], u[5], u[6], u[7]);
}
__global__ void k_pack(const float* w1, const float* w2, const float* wblk,
                       uint4* p1, uint4* p2, uint4* pb) {
    int gid = blockIdx.x * blockDim.x + threadIdx.x;
    const int NC = 18 * 4 * 32;
    if (gid < NC)                 pack_conv(w1, p1, gid);
    else if (gid < 2 * NC)        pack_conv(w2, p2, gid - NC);
    else if (gid < 2 * NC + 256)  pack_mat64(wblk, pb, gid - 2 * NC);
}

// ---------------- 3x3 conv via implicit-GEMM WMMA ----------------
// MODE 0: input f32 NHWC (res), no residual  -> out = relu(acc*g+b)
// MODE 1: input bf16 NHWC (h1), residual res -> out = relu(acc*g+b+res)
// Fully-unrolled constant-trip tap loops: 18 static WMMAs, branchless
// boundary handling (clamped address + cndmask zero) keeps EXEC full and
// lets the compiler clause the global loads and fill WMMA hazard slots.
template <int MODE>
__global__ void k_conv3x3(const float* __restrict__ inF32,
                          const unsigned short* __restrict__ inBf16,
                          const float* __restrict__ resNHWC,
                          const uint4* __restrict__ wpack,
                          const float* __restrict__ gamma, const float* __restrict__ beta,
                          unsigned short* __restrict__ outBf16) {
    int tid = threadIdx.x;
    int wave = tid >> 5, lane = tid & 31;
    int t = blockIdx.x;
    int xt = t % XT;
    int y = (t / XT) % OBJ;
    int b = t / (XT * OBJ);
    int xbase = xt * 16;
    int m = lane & 15, laneHi = lane >> 4;

    v8f acc = {0.f, 0.f, 0.f, 0.f, 0.f, 0.f, 0.f, 0.f};
#pragma unroll
    for (int ky = 0; ky < 3; ++ky) {
#pragma unroll
        for (int kx = 0; kx < 3; ++kx) {
            int srcy = y + ky - 1;
            int srcx = xbase + m + kx - 1;
            bool inb = (srcy >= 0) && (srcy < OBJ) && (srcx >= 0) && (srcx < OBJ);
            int cy = srcy < 0 ? 0 : (srcy > OBJ - 1 ? OBJ - 1 : srcy);
            int cx = srcx < 0 ? 0 : (srcx > OBJ - 1 ? OBJ - 1 : srcx);
            size_t pix = ((size_t)(b * OBJ + cy) * OBJ + cx) * 64;
#pragma unroll
            for (int c2 = 0; c2 < 2; ++c2) {
                const int kc = (ky * 3 + kx) * 2 + c2;
                int ic0 = c2 * 32 + laneHi * 8;
                FragBF a;
                if (MODE == 0) {
                    const float4* p0 = (const float4*)(inF32 + pix + ic0);
                    const float4* p1 = (const float4*)(inF32 + pix + ic0 + 16);
                    float4 x0 = p0[0], x1 = p0[1];
                    float4 x2 = p1[0], x3 = p1[1];
                    a.u[0] = bf16_pack2(x0.x, x0.y);
                    a.u[1] = bf16_pack2(x0.z, x0.w);
                    a.u[2] = bf16_pack2(x1.x, x1.y);
                    a.u[3] = bf16_pack2(x1.z, x1.w);
                    a.u[4] = bf16_pack2(x2.x, x2.y);
                    a.u[5] = bf16_pack2(x2.z, x2.w);
                    a.u[6] = bf16_pack2(x3.x, x3.y);
                    a.u[7] = bf16_pack2(x3.z, x3.w);
                } else {
                    a.q[0] = *(const uint4*)(inBf16 + pix + ic0);
                    a.q[1] = *(const uint4*)(inBf16 + pix + ic0 + 16);
                }
                // branchless zero for out-of-bounds taps (v_cndmask, no EXEC games)
#pragma unroll
                for (int v = 0; v < 8; ++v) a.u[v] = inb ? a.u[v] : 0u;

                const uint4* wp = wpack + ((size_t)(kc * 4 + wave) * 32 + lane) * 2;
                FragBF bf;
                bf.q[0] = wp[0];
                bf.q[1] = wp[1];
                acc = wmma_bf16(a.v, bf.v, acc);
            }
        }
    }
    // epilogue: C/D layout -> lane = channel (within octile), VGPR r = pixel M (+8 for hi lanes)
    int ch = wave * 16 + (lane & 15);
    float gm = gamma[ch], bt = beta[ch];
#pragma unroll
    for (int r = 0; r < 8; ++r) {
        int mm = r + (laneHi ? 8 : 0);
        size_t o = ((size_t)(b * OBJ + y) * OBJ + (xbase + mm)) * 64 + ch;
        float v = acc[r] * gm + bt;
        if (MODE == 1) v += resNHWC[o];
        v = v > 0.f ? v : 0.f;
        outBf16[o] = (unsigned short)bf16_bits(v);
    }
}

// ---------------- 1x1 (64->64) WMMA + image head (64->3) ----------------
__global__ void k_out(const unsigned short* __restrict__ h2,
                      const uint4* __restrict__ wpackBlk,
                      const float* __restrict__ b_blkout,
                      const float* __restrict__ w_img, const float* __restrict__ b_img,
                      float* __restrict__ out) {
    __shared__ float sh[16][64];
    int tid = threadIdx.x;
    int wave = tid >> 5, lane = tid & 31;
    int t = blockIdx.x;
    int xt = t % XT;
    int y = (t / XT) % OBJ;
    int b = t / (XT * OBJ);
    int xbase = xt * 16;
    int m = lane & 15, laneHi = lane >> 4;

    v8f acc = {0.f, 0.f, 0.f, 0.f, 0.f, 0.f, 0.f, 0.f};
#pragma unroll
    for (int kc = 0; kc < 2; ++kc) {
        int ic0 = kc * 32 + laneHi * 8;
        size_t pix = ((size_t)(b * OBJ + y) * OBJ + (xbase + m)) * 64;
        FragBF a;
        a.q[0] = *(const uint4*)(h2 + pix + ic0);
        a.q[1] = *(const uint4*)(h2 + pix + ic0 + 16);
        const uint4* wp = wpackBlk + ((size_t)(kc * 4 + wave) * 32 + lane) * 2;
        FragBF bf;
        bf.q[0] = wp[0];
        bf.q[1] = wp[1];
        acc = wmma_bf16(a.v, bf.v, acc);
    }
    int ch = wave * 16 + (lane & 15);
    float bb = b_blkout[ch];
#pragma unroll
    for (int r = 0; r < 8; ++r) {
        int mm = r + (laneHi ? 8 : 0);
        sh[mm][ch] = acc[r] + bb;
    }
    __syncthreads();
    if (tid < 48) {
        int mm = tid / 3, o = tid % 3;
        const float mean3[3] = {0.485f, 0.456f, 0.406f};
        const float std3[3]  = {0.229f, 0.224f, 0.225f};
        float s = b_img[o];
        for (int c = 0; c < 64; ++c) s += sh[mm][c] * w_img[o * 64 + c];
        float sig = 1.f / (1.f + expf(-s));
        out[((size_t)(b * 3 + o) * OBJ + y) * OBJ + (xbase + mm)] = (sig - mean3[o]) / std3[o];
    }
}

// ---------------- launch ----------------
extern "C" void kernel_launch(void* const* d_in, const int* in_sizes, int n_in,
                              void* d_out, int out_size, void* d_ws, size_t ws_size,
                              hipStream_t stream) {
    (void)in_sizes; (void)n_in; (void)out_size; (void)ws_size;
    const float* pts      = (const float*)d_in[0];
    const float* w_in     = (const float*)d_in[1];
    const float* b_in     = (const float*)d_in[2];
    const float* w_graph  = (const float*)d_in[3];
    const float* gn_g     = (const float*)d_in[4];
    const float* gn_b     = (const float*)d_in[5];
    const float* w_proj   = (const float*)d_in[6];
    const float* b_proj   = (const float*)d_in[7];
    const float* bb_w1    = (const float*)d_in[8];
    const float* bb_g1    = (const float*)d_in[9];
    const float* bb_b1    = (const float*)d_in[10];
    const float* bb_w2    = (const float*)d_in[11];
    const float* bb_g2    = (const float*)d_in[12];
    const float* bb_b2    = (const float*)d_in[13];
    const float* w_blkout = (const float*)d_in[14];
    const float* b_blkout = (const float*)d_in[15];
    const float* w_img    = (const float*)d_in[16];
    const float* b_img    = (const float*)d_in[17];

    char* ws = (char*)d_ws;
    float*          geom   = (float*)(ws + OFF_GEOM);
    float*          gnsum  = (float*)(ws + OFF_GNSUM);
    float*          gnstat = (float*)(ws + OFF_GNSTAT);
    float*          f0     = (float*)(ws + OFF_F0);
    int*            knn    = (int*)  (ws + OFF_KNN);
    float*          g      = (float*)(ws + OFF_G);
    float*          res    = (float*)(ws + OFF_RES);
    unsigned short* h1     = (unsigned short*)(ws + OFF_H1);
    unsigned short* h2     = (unsigned short*)(ws + OFF_H2);
    uint4*          p1     = (uint4*)(ws + OFF_P1);
    uint4*          p2     = (uint4*)(ws + OFF_P2);
    uint4*          pb     = (uint4*)(ws + OFF_PB);

    const int pixBlocks = Bn * OBJ * XT;                       // 25088

    // zero res grid + GN sums (ws is poisoned, must init)
    k_zero<<<pixBlocks, 256, 0, stream>>>((float4*)res, (long long)Bn * OBJ * OBJ * 64 / 4);
    k_zero<<<1, 256, 0, stream>>>((float4*)gnsum, 16LL);

    k_geom <<<Bn, 256, 0, stream>>>(pts, geom);
    k_knn  <<<Bn * (Npt / 256), 256, 0, stream>>>(pts, w_in, b_in, f0, knn);
    k_graph<<<(Bn * Npt * Kk) / 256, 256, 0, stream>>>(f0, knn, w_graph, g, gnsum);
    k_gnfin<<<1, 32, 0, stream>>>(gnsum, gnstat);
    k_fproj<<<(Bn * Npt) / 256, 256, 0, stream>>>(g, gnstat, gn_g, gn_b, w_proj, b_proj,
                                                  pts, geom, res);
    k_pack <<<19, 256, 0, stream>>>(bb_w1, bb_w2, w_blkout, p1, p2, pb);

    k_conv3x3<0><<<pixBlocks, 128, 0, stream>>>(res, nullptr, nullptr, p1, bb_g1, bb_b1, h1);
    k_conv3x3<1><<<pixBlocks, 128, 0, stream>>>(nullptr, h1, res, p2, bb_g2, bb_b2, h2);
    k_out      <<<pixBlocks, 128, 0, stream>>>(h2, pb, b_blkout, w_img, b_img, (float*)d_out);
}
